// SparNetLSTM_82300163326794
// MI455X (gfx1250) — matrix operand order repaired and checked by reading, not compiled, not run on hardware
//
#include <hip/hip_runtime.h>
#include <hip/hip_bf16.h>

typedef __attribute__((ext_vector_type(16))) _Float16 v16h;
typedef __attribute__((ext_vector_type(8)))  _Float16 v8h;
typedef __attribute__((ext_vector_type(4)))  _Float16 v4h;
typedef __attribute__((ext_vector_type(8)))  float    v8f;

// ---------------- constants ----------------
#define SLEN 1000
#define BATCH 64
#define DIMS 256
#define MROWS (SLEN * BATCH)                  // 64000
#define GATE_STRIDE ((size_t)MROWS * DIMS)    // elements per gate
#define GX_DIR_STRIDE ((size_t)MROWS * DIMS * 4)  // halves per direction (4 gates interleaved)

// workspace layout (bytes)
static constexpr size_t SZ_TSEQ = (size_t)MROWS * DIMS * 2;        // 32,768,000
static constexpr size_t SZ_W    = 8ull * 256 * 256 * 2;            // 1,048,576
static constexpr size_t OFF_TSEQ = 0;
static constexpr size_t OFF_WH   = OFF_TSEQ + SZ_TSEQ;             // WhT [8][256K][256N]
static constexpr size_t OFF_WX   = OFF_WH + SZ_W;                  // WxT [8][256K][256N]
static constexpr size_t OFF_B16  = OFF_WX + SZ_W;                  // 8*256*2 = 4096
static constexpr size_t OFF_WY   = OFF_B16 + 4096;                 // WyT [512K][128N] = 131072
static constexpr size_t OFF_GX   = OFF_WY + 131072;                // [2][64000][256][4] f16
static constexpr size_t OFF_Y    = OFF_GX + 8ull * GATE_STRIDE * 2;
static constexpr size_t OFF_STAT = OFF_Y + (size_t)MROWS * 512 * 2;

// ---------------- WMMA helpers ----------------
__device__ __forceinline__ v8f wmma32(v16h a, v16h b, v8f c) {
    return __builtin_amdgcn_wmma_f32_16x16x32_f16(false, a, false, b, (short)0, c, false, false);
}

__device__ __forceinline__ v16h cat8(v8h lo, v8h hi) {
    return __builtin_shufflevector(lo, hi, 0, 1, 2, 3, 4, 5, 6, 7,
                                          8, 9, 10, 11, 12, 13, 14, 15);
}

// A tile 16x32 (MxK), row-major source with leading dim ld (halves).
// lane = M (mod 16); lanes<16 hold K 0..7,16..23 ; lanes>=16 hold K 8..15,24..31
// => two contiguous 16-byte chunks per lane (b128 loads).
__device__ __forceinline__ v16h load_fragA(const _Float16* base, int ld) {
    int lane = threadIdx.x & 31;
    int m = lane & 15;
    int khi = (lane < 16) ? 0 : 8;
    const _Float16* p = base + (size_t)m * ld + khi;
    v8h lo = *(const v8h*)(p);
    v8h hi = *(const v8h*)(p + 16);
    return cat8(lo, hi);
}

// B tile 32x16 (KxN) from K-major (pre-transposed) weights: wt points at (k0,n0),
// leading dim ld (halves). lane = N (mod 16); lanes<16 hold K 0..7,16..23 ; lanes>=16 hold K 8..15,24..31.
__device__ __forceinline__ v16h load_fragB_t(const _Float16* wt, int ld) {
    int lane = threadIdx.x & 31;
    int n = lane & 15;
    int khi = (lane < 16) ? 0 : 8;
    const _Float16* p = wt + (size_t)khi * ld + n;
    v16h r;
#pragma unroll
    for (int s = 0; s < 8; ++s) {
        r[s] = p[(size_t)s * ld];
        r[8 + s] = p[(size_t)(16 + s) * ld];
    }
    return r;
}

__device__ __forceinline__ float sigmoidf(float x) { return 1.0f / (1.0f + __expf(-x)); }

// ---------------- weight prep: f32 -> f16, split + transpose to K-major ----------------
struct GatePtrs { const float* W[8]; const float* b[8]; };

__global__ __launch_bounds__(256)
void k_wprep(GatePtrs gp, const float* Wy, _Float16* WhT, _Float16* WxT,
             _Float16* b16, _Float16* WyT) {
    int idx = blockIdx.x * 256 + threadIdx.x;
    if (idx < 8 * 256 * 512) {
        int g = idx / (256 * 512);
        int r = idx % (256 * 512);
        int n = r / 512, k = r % 512;
        _Float16 v = (_Float16)gp.W[g][(size_t)n * 512 + k];
        if (k < 256) WhT[(size_t)g * 65536 + (size_t)k * 256 + n] = v;
        else         WxT[(size_t)g * 65536 + (size_t)(k - 256) * 256 + n] = v;
    } else if (idx < 8 * 256 * 512 + 2048) {
        int r = idx - 8 * 256 * 512;
        int g = r / 256, j = r % 256;
        b16[r] = (_Float16)gp.b[g][j];
    } else if (idx < 8 * 256 * 512 + 2048 + 512 * 128) {
        int r = idx - (8 * 256 * 512 + 2048);
        int k = r / 128, n = r % 128;
        WyT[r] = (n < 95) ? (_Float16)Wy[(size_t)n * 512 + k] : (_Float16)0.0f;
    }
}

// ---------------- conv 5x5 (pad 2) + bias + maxpool 2x2 (pad 1) -> t_seq f16 ----------------
__global__ __launch_bounds__(256)
void k_conv_pool(const float* __restrict__ x, const float* __restrict__ cw,
                 const float* __restrict__ cb, _Float16* __restrict__ t_seq) {
    int idx = blockIdx.x * 256 + threadIdx.x;       // 64*16*1000*16
    int w = idx & 15;
    int s = (idx >> 4) % 1000;
    int c = (idx / 16000) & 15;
    int b = idx / 256000;
    const float* xb = x + (size_t)b * 1998 * 30;
    const float* wc = cw + c * 25;
    float bias = cb[c];
    float best = -3.4e38f;
#pragma unroll
    for (int ph = 0; ph < 2; ++ph) {
        int hh = 2 * s - 1 + ph;
        if (hh < 0 || hh >= 1998) continue;
#pragma unroll
        for (int pw = 0; pw < 2; ++pw) {
            int ww = 2 * w - 1 + pw;
            if (ww < 0 || ww >= 30) continue;
            float acc = bias;
#pragma unroll
            for (int kh = 0; kh < 5; ++kh) {
                int row = hh - 2 + kh;
                if (row < 0 || row >= 1998) continue;
#pragma unroll
                for (int kw = 0; kw < 5; ++kw) {
                    int col = ww - 2 + kw;
                    if (col < 0 || col >= 30) continue;
                    acc += xb[(size_t)row * 30 + col] * wc[kh * 5 + kw];
                }
            }
            best = fmaxf(best, acc);
        }
    }
    t_seq[((size_t)s * 64 + b) * 256 + c * 16 + w] = (_Float16)best;
}

// ---------------- input projections -> gates_x[d][row][col][gate] (interleaved) ----------------
__global__ __launch_bounds__(128)
void k_xgemm(const _Float16* __restrict__ t_seq, const _Float16* __restrict__ WxT,
             const _Float16* __restrict__ b16, _Float16* __restrict__ gates_x) {
    int mtile = blockIdx.x;           // 0..3999
    int g     = blockIdx.y;           // 0..7
    int wave  = threadIdx.x >> 5;     // 0..3
    int lane  = threadIdx.x & 31;
    int d     = g >> 2;               // direction
    int gl    = g & 3;                // gate within direction

    const _Float16* Arow = t_seq + (size_t)mtile * 16 * 256;
    v16h Af[8];
#pragma unroll
    for (int kf = 0; kf < 8; ++kf) Af[kf] = load_fragA(Arow + kf * 32, 256);

    const _Float16* Wg = WxT + (size_t)g * 65536;       // [K=256][N=256]
    _Float16* outd = gates_x + (size_t)d * GX_DIR_STRIDE;

#pragma unroll
    for (int i = 0; i < 4; ++i) {
        int nt = wave * 4 + i;
        v8f acc = {};
#pragma unroll
        for (int kf = 0; kf < 8; ++kf) {
            v16h Bf = load_fragB_t(Wg + (size_t)(kf * 32) * 256 + nt * 16, 256);
            acc = wmma32(Af[kf], Bf, acc);
        }
        int col = nt * 16 + (lane & 15);
        float bias = (float)b16[g * 256 + col];
#pragma unroll
        for (int r = 0; r < 8; ++r) {
            int m = mtile * 16 + r + ((lane >= 16) ? 8 : 0);
            outd[((size_t)m * 256 + col) * 4 + gl] = (_Float16)(acc[r] + bias);
        }
    }
}

// ---------------- persistent bidirectional LSTM recurrence ----------------
// grid = 2 blocks (one per direction), 512 threads = 16 waves; wave w owns N-tile w.
__global__ __launch_bounds__(512)
void k_lstm(const _Float16* __restrict__ gates_x, const _Float16* __restrict__ WhT,
            const float* __restrict__ c1, const float* __restrict__ c2,
            _Float16* __restrict__ y) {
    __shared__ alignas(16) _Float16 h_lds[64][256];
    const int d    = blockIdx.x;
    const int wave = threadIdx.x >> 5;   // = ntile
    const int lane = threadIdx.x & 31;
    const int j    = wave * 16 + (lane & 15);

    const float c0 = (d ? c2 : c1)[j];
    float c_reg[4][8];
#pragma unroll
    for (int mt = 0; mt < 4; ++mt)
#pragma unroll
        for (int r = 0; r < 8; ++r) c_reg[mt][r] = c0;

    const _Float16 h0 = (_Float16)tanhf(c0);
    for (int b = (lane >> 4); b < 64; b += 2) h_lds[b][j] = h0;
    __syncthreads();

    const _Float16* Whd = WhT + (size_t)d * 4 * 65536;  // [gate][K=256][N=256]
    const _Float16* gxd = gates_x + (size_t)d * GX_DIR_STRIDE;  // [row][col][gate]

    for (int t = 0; t < SLEN; ++t) {
        const int s = d ? (SLEN - 1 - t) : t;
        // prefetch next step's interleaved gate block (128 KB) into L2
        if (t + 1 < SLEN) {
            int sn = d ? (SLEN - 2 - t) : (t + 1);
            __builtin_prefetch(gxd + (size_t)sn * 64 * 1024 + (size_t)threadIdx.x * 128, 0, 1);
        }

        v8f acc[4][4];   // [gate][mtile]
#pragma unroll
        for (int g = 0; g < 4; ++g)
#pragma unroll
            for (int mt = 0; mt < 4; ++mt) acc[g][mt] = (v8f){};

#pragma unroll
        for (int kf = 0; kf < 8; ++kf) {
            v16h Bf[4];
#pragma unroll
            for (int g = 0; g < 4; ++g)
                Bf[g] = load_fragB_t(Whd + (size_t)g * 65536 + (size_t)(kf * 32) * 256 + wave * 16, 256);
#pragma unroll
            for (int mt = 0; mt < 4; ++mt) {
                v16h Af = load_fragA(&h_lds[mt * 16][0] + kf * 32, 256);
#pragma unroll
                for (int g = 0; g < 4; ++g) acc[g][mt] = wmma32(Af, Bf[g], acc[g][mt]);
            }
        }
        __syncthreads();   // all waves done reading h_lds

#pragma unroll
        for (int mt = 0; mt < 4; ++mt) {
#pragma unroll
            for (int r = 0; r < 8; ++r) {
                int b = mt * 16 + r + ((lane >= 16) ? 8 : 0);
                size_t row = ((size_t)s * 64 + b) * 256 + j;
                v4h gv = *(const v4h*)(gxd + row * 4);     // one b64: i,f,o,g
                float pi = acc[0][mt][r] + (float)gv[0];
                float pf = acc[1][mt][r] + (float)gv[1];
                float po = acc[2][mt][r] + (float)gv[2];
                float pg = acc[3][mt][r] + (float)gv[3];
                float ig = sigmoidf(pi), fg = sigmoidf(pf), og = sigmoidf(po);
                float gg = tanhf(pg);
                float c = ig * gg + fg * c_reg[mt][r];
                c_reg[mt][r] = c;
                _Float16 hh = (_Float16)(og * tanhf(c));
                h_lds[b][j] = hh;
                y[((size_t)s * 64 + b) * 512 + d * 256 + j] = hh;
            }
        }
        __syncthreads();   // h_lds updated for next step
    }
}

// ---------------- per-feature stats over (S,B) ----------------
__global__ __launch_bounds__(256)
void k_stats(const _Float16* __restrict__ y, float* __restrict__ mean, float* __restrict__ inv) {
    __shared__ float sb[256], qb[256];
    int j = blockIdx.x;          // 0..511
    float s = 0.f, q = 0.f;
    for (int m = threadIdx.x; m < MROWS; m += 256) {
        float v = (float)y[(size_t)m * 512 + j];
        s += v; q += v * v;
    }
    sb[threadIdx.x] = s; qb[threadIdx.x] = q;
    __syncthreads();
    for (int off = 128; off > 0; off >>= 1) {
        if (threadIdx.x < off) {
            sb[threadIdx.x] += sb[threadIdx.x + off];
            qb[threadIdx.x] += qb[threadIdx.x + off];
        }
        __syncthreads();
    }
    if (threadIdx.x == 0) {
        float mu = sb[0] / (float)MROWS;
        float var = qb[0] / (float)MROWS - mu * mu;
        mean[j] = mu;
        inv[j] = rsqrtf(var + 1e-5f);
    }
}

// ---------------- normalize in place ----------------
__global__ __launch_bounds__(256)
void k_norm(_Float16* __restrict__ y, const float* __restrict__ mean, const float* __restrict__ inv) {
    size_t idx = (size_t)blockIdx.x * 256 + threadIdx.x;  // 64000*512
    int j = (int)(idx & 511);
    y[idx] = (_Float16)(((float)y[idx] - mean[j]) * inv[j]);
}

// ---------------- final projection: out[b][s][v] = yn @ Wy^T ----------------
__global__ __launch_bounds__(192)
void k_ygemm(const _Float16* __restrict__ yn, const _Float16* __restrict__ WyT,
             float* __restrict__ out) {
    int mtile = blockIdx.x;          // 0..3999
    int nt    = threadIdx.x >> 5;    // 0..5
    int lane  = threadIdx.x & 31;
    const _Float16* Arow = yn + (size_t)mtile * 16 * 512;
    v8f acc = {};
#pragma unroll
    for (int kf = 0; kf < 16; ++kf) {
        v16h Af = load_fragA(Arow + kf * 32, 512);
        v16h Bf = load_fragB_t(WyT + (size_t)(kf * 32) * 128 + nt * 16, 128);
        acc = wmma32(Af, Bf, acc);
    }
    int v = nt * 16 + (lane & 15);
    if (v < 95) {
#pragma unroll
        for (int r = 0; r < 8; ++r) {
            int m = mtile * 16 + r + ((lane >= 16) ? 8 : 0);
            int s = m / 64, b = m % 64;
            out[((size_t)b * SLEN + s) * 95 + v] = acc[r];
        }
    }
}

// ---------------- host launcher ----------------
extern "C" void kernel_launch(void* const* d_in, const int* in_sizes, int n_in,
                              void* d_out, int out_size, void* d_ws, size_t ws_size,
                              hipStream_t stream) {
    (void)in_sizes; (void)n_in; (void)out_size; (void)ws_size;
    const float* x      = (const float*)d_in[0];
    const float* conv_w = (const float*)d_in[1];
    const float* conv_b = (const float*)d_in[2];
    GatePtrs gp;
    for (int g = 0; g < 8; ++g) {
        gp.W[g] = (const float*)d_in[3 + 2 * g];
        gp.b[g] = (const float*)d_in[4 + 2 * g];
    }
    const float* c1  = (const float*)d_in[19];
    const float* c2  = (const float*)d_in[20];
    const float* W_y = (const float*)d_in[21];

    char* ws = (char*)d_ws;
    _Float16* t_seq   = (_Float16*)(ws + OFF_TSEQ);
    _Float16* WhT     = (_Float16*)(ws + OFF_WH);
    _Float16* WxT     = (_Float16*)(ws + OFF_WX);
    _Float16* b16     = (_Float16*)(ws + OFF_B16);
    _Float16* WyT     = (_Float16*)(ws + OFF_WY);
    _Float16* gates_x = (_Float16*)(ws + OFF_GX);
    _Float16* ybuf    = (_Float16*)(ws + OFF_Y);
    float*    mean    = (float*)(ws + OFF_STAT);
    float*    inv     = mean + 512;
    float*    out     = (float*)d_out;

    // 1. weight conversion / split / transpose (K-major [K][N] layout)
    k_wprep<<<dim3((8 * 256 * 512 + 2048 + 512 * 128 + 255) / 256), dim3(256), 0, stream>>>(
        gp, W_y, WhT, WxT, b16, WyT);
    // 2. conv + pool -> sequence (f16)
    k_conv_pool<<<dim3(64000), dim3(256), 0, stream>>>(x, conv_w, conv_b, t_seq);
    // 3. batched input projections for all 8 gates (WMMA), gate-interleaved output
    k_xgemm<<<dim3(4000, 8), dim3(128), 0, stream>>>(t_seq, WxT, b16, gates_x);
    // 4. persistent bidirectional recurrence (WMMA, h in LDS)
    k_lstm<<<dim3(2), dim3(512), 0, stream>>>(gates_x, WhT, c1, c2, ybuf);
    // 5. feature-wise mean / rsqrt(var)
    k_stats<<<dim3(512), dim3(256), 0, stream>>>(ybuf, mean, inv);
    // 6. normalize in place
    k_norm<<<dim3((MROWS * 512) / 256), dim3(256), 0, stream>>>(ybuf, mean, inv);
    // 7. output projection (WMMA) with transpose store
    k_ygemm<<<dim3(4000), dim3(192), 0, stream>>>(ybuf, WyT, out);
}
